// HierarchicalMalwareGNN_39058432590504
// MI455X (gfx1250) — matrix operand order repaired and checked by reading, not compiled, not run on hardware
//
#include <hip/hip_runtime.h>
#include <hip/hip_bf16.h>

// ---------------------------------------------------------------------------
// HierarchicalMalwareGNN inference for MI455X (gfx1250, wave32, WMMA)
// Big node-matrix GEMMs: LDS-staged weights (bf16 pairs, ds_load_b128) with
// one wave per 16-row strip sweeping all column tiles -> 16 WMMAs per A-frag.
// Edge aggregation / pooling use global f32 atomics (bandwidth-bound).
// ---------------------------------------------------------------------------

#define NNODES 100000
#define NEDGES 1000000
#define NB     64
#define DIN    128
#define DS     1024
#define HDIM   256
#define NG     20

typedef __attribute__((ext_vector_type(16))) __bf16 v16bf;
typedef __attribute__((ext_vector_type(8)))  float  v8f;

union BFrag { v16bf v; unsigned int u[8]; };

__device__ __forceinline__ unsigned int pack_bf16(float lo, float hi) {
    union { __bf16 h[2]; unsigned int u; } p;
    p.h[0] = (__bf16)lo;
    p.h[1] = (__bf16)hi;
    return p.u;
}

// ---------------- elementwise / utility kernels ----------------

__global__ void k_fill(float* __restrict__ p, float v, long long n) {
    long long i = (long long)blockIdx.x * blockDim.x + threadIdx.x;
    if (i < n) p[i] = v;
}

__global__ void k_degree(const int* __restrict__ dst, float* __restrict__ deg, int e) {
    int i = blockIdx.x * blockDim.x + threadIdx.x;
    if (i < e) atomicAdd(&deg[dst[i]], 1.0f);
}

__global__ void k_dinv(const float* __restrict__ deg, float* __restrict__ dinv, int n) {
    int i = blockIdx.x * blockDim.x + threadIdx.x;
    if (i < n) dinv[i] = rsqrtf(deg[i]);
}

__global__ void k_edgew(const int* __restrict__ src, const int* __restrict__ dst,
                        const float* __restrict__ dinv, float* __restrict__ ew, int e) {
    int i = blockIdx.x * blockDim.x + threadIdx.x;
    if (i < e) ew[i] = dinv[src[i]] * dinv[dst[i]];
}

// agg[dst] += h[src] * ew  (one thread per edge-feature)
__global__ void k_scatter(const int* __restrict__ src, const int* __restrict__ dst,
                          const float* __restrict__ ew, const float* __restrict__ P,
                          float* __restrict__ AGG, long long total) {
    long long t = (long long)blockIdx.x * blockDim.x + threadIdx.x;
    if (t >= total) return;
    long long e = t >> 8;            // /HDIM
    int j = (int)(t & (HDIM - 1));
    int s = src[e], d = dst[e];
    atomicAdd(&AGG[(long long)d * HDIM + j], P[(long long)s * HDIM + j] * ew[e]);
}

// Xout = relu(AGG + P/deg + bc) + R
__global__ void k_combine(const float* __restrict__ AGG, const float* __restrict__ P,
                          const float* __restrict__ R, const float* __restrict__ deg,
                          const float* __restrict__ bc, float* __restrict__ X,
                          long long total) {
    long long t = (long long)blockIdx.x * blockDim.x + threadIdx.x;
    if (t >= total) return;
    long long i = t >> 8;
    int j = (int)(t & (HDIM - 1));
    float v = AGG[t] + P[t] / deg[i] + bc[j];
    X[t] = fmaxf(v, 0.0f) + R[t];
}

__global__ void k_counts(const int* __restrict__ batch, float* __restrict__ counts, int n) {
    int i = blockIdx.x * blockDim.x + threadIdx.x;
    if (i < n) atomicAdd(&counts[batch[i]], 1.0f);
}

// pool x3 -> combined[:, 0:256], pat -> combined[:, 1024:1280]
__global__ void k_pool(const int* __restrict__ batch, const float* __restrict__ X3,
                       const float* __restrict__ PAT, float* __restrict__ comb,
                       long long total) {
    long long t = (long long)blockIdx.x * blockDim.x + threadIdx.x;
    if (t >= total) return;
    long long i = t >> 8;
    int j = (int)(t & (HDIM - 1));
    int b = batch[i];
    atomicAdd(&comb[(long long)b * 1280 + j],        X3[t]);
    atomicAdd(&comb[(long long)b * 1280 + 1024 + j], PAT[t]);
}

__global__ void k_pooldiv(const float* __restrict__ counts, float* __restrict__ comb, int total) {
    int t = blockIdx.x * blockDim.x + threadIdx.x;
    if (t >= total) return;
    int b = t >> 8;
    int j = t & (HDIM - 1);
    float inv = 1.0f / fmaxf(counts[b], 1.0f);
    comb[(long long)b * 1280 + j]        *= inv;
    comb[(long long)b * 1280 + 1024 + j] *= inv;
}

// temporal layer 1: [B,5] @ [5,128] + b, relu  (K=5, scalar)
__global__ void k_temporal1(const float* __restrict__ tf, const float* __restrict__ W,
                            const float* __restrict__ b, float* __restrict__ out, int total) {
    int t = blockIdx.x * blockDim.x + threadIdx.x;
    if (t >= total) return;
    int bi = t >> 7;                 // /128
    int j  = t & 127;
    float s = b[j];
    #pragma unroll
    for (int k = 0; k < 5; ++k) s += tf[bi * 5 + k] * W[k * 128 + j];
    out[t] = fmaxf(s, 0.0f);
}

__global__ void k_typegather(const int* __restrict__ idx, const float* __restrict__ emb,
                             float* __restrict__ comb, int total) {
    int t = blockIdx.x * blockDim.x + threadIdx.x;
    if (t >= total) return;
    int b = t >> 8;
    int j = t & (HDIM - 1);
    comb[(long long)b * 1280 + 768 + j] = emb[(long long)idx[b] * HDIM + j];
}

// group head: [B,256] @ [256,20] + bg  (Nc=20, scalar)
__global__ void k_grouphead(const float* __restrict__ fused, const float* __restrict__ Wg,
                            const float* __restrict__ bg, float* __restrict__ out, int total) {
    int t = blockIdx.x * blockDim.x + threadIdx.x;
    if (t >= total) return;
    int b = t / NG;
    int j = t % NG;
    float s = bg[j];
    for (int k = 0; k < HDIM; ++k) s += fused[b * HDIM + k] * Wg[k * NG + j];
    out[t] = s;
}

// time head: [B,256] @ [256,1] + btp
__global__ void k_timehead(const float* __restrict__ fused, const float* __restrict__ Wtp,
                           const float* __restrict__ btp, float* __restrict__ out, int nb) {
    int b = blockIdx.x * blockDim.x + threadIdx.x;
    if (b >= nb) return;
    float s = btp[0];
    for (int k = 0; k < HDIM; ++k) s += fused[b * HDIM + k] * Wtp[k];
    out[b] = s;
}

// ---------------- strip WMMA GEMM for large-M GEMMs ----------------
// C[M,NC] = A[M,K] @ B[K,NC] (+bias)(+relu), K/NC compile-time.
// Block = 256 threads (8 waves). The whole weight matrix is staged in
// (dynamic) LDS as bf16 pairs, column-major with +4 dword pad so a B
// fragment is 8 contiguous, 16B-aligned LDS dwords (2x ds_load_b128) and
// lanes stride 4 banks. Each wave owns one 16-row strip of A and sweeps
// all NC/16 column tiles: one A fragment feeds NC/16 consecutive WMMAs.

template <int K, int NC>
__global__ void k_gemm_strip(const float* __restrict__ A, const float* __restrict__ Bm,
                             const float* __restrict__ bias, float* __restrict__ C,
                             int M, int ldc, int relu) {
    constexpr int NTILES = NC / 16;
    constexpr int KP2 = K / 2;
    constexpr int STR = KP2 + 4;          // dword stride per column (16B aligned, bank-skewed)
    extern __shared__ unsigned int sB[];  // NC * STR dwords

    // cooperative stage: pack B[2kp][col], B[2kp+1][col] -> one dword
    for (int idx = threadIdx.x; idx < NC * KP2; idx += blockDim.x) {
        int col = idx / KP2;
        int kp  = idx - col * KP2;
        sB[col * STR + kp] =
            pack_bf16(Bm[(long long)(2 * kp) * NC + col],
                      Bm[(long long)(2 * kp + 1) * NC + col]);
    }
    __syncthreads();

    int lane = threadIdx.x & 31;
    int wave = threadIdx.x >> 5;
    int strip = blockIdx.x * (blockDim.x >> 5) + wave;
    if (strip >= (M >> 4)) return;
    int tm = strip << 4;

    int half = lane >> 4;                 // 0: lanes 0-15, 1: lanes 16-31
    int l15  = lane & 15;
    long long arow = tm + l15;
    int kA = half * 16;                   // A: half0 K0..15, half1 K16..31
    int kB = half * 16;                   // B: half0 K0..15, half1 K16..31

    v8f acc[NTILES] = {};

    for (int kk = 0; kk < K; kk += 32) {
        v16bf a;
        const float* ap = A + arow * K + kk + kA;
        #pragma unroll
        for (int i = 0; i < 8; ++i) {
            a[i]     = (__bf16)ap[i];
            a[i + 8] = (__bf16)ap[8 + i];
        }
        #pragma unroll
        for (int t = 0; t < NTILES; ++t) {
            BFrag bf;
            const unsigned int* bp = sB + (t * 16 + l15) * STR + ((kk + kB) >> 1);
            #pragma unroll
            for (int i = 0; i < 8; ++i) bf.u[i] = bp[i];
            acc[t] = __builtin_amdgcn_wmma_f32_16x16x32_bf16(
                false, a, false, bf.v, (short)0, acc[t], false, false);
        }
    }

    #pragma unroll
    for (int t = 0; t < NTILES; ++t) {
        int col = t * 16 + l15;
        float bv = bias ? bias[col] : 0.0f;
        #pragma unroll
        for (int r = 0; r < 8; ++r) {
            long long row = tm + half * 8 + r;
            float v = acc[t][r] + bv;
            if (relu) v = fmaxf(v, 0.0f);
            C[row * (long long)ldc + col] = v;
        }
    }
}

// ---------------- generic per-tile WMMA GEMM (small M / big K weights) ----------------

__global__ void k_gemm_wmma(const float* __restrict__ A, const float* __restrict__ Bm,
                            const float* __restrict__ bias, float* __restrict__ C,
                            int M, int K, int Nc, int ldc, int relu) {
    int lane = threadIdx.x & 31;
    int wave = threadIdx.x >> 5;
    int tilesN = Nc >> 4;
    int ntiles = (M >> 4) * tilesN;
    int tile = blockIdx.x * (blockDim.x >> 5) + wave;
    if (tile >= ntiles) return;
    int tm = (tile / tilesN) << 4;
    int tn = (tile % tilesN) << 4;

    int half = lane >> 4;
    int l15  = lane & 15;
    long long arow = tm + l15;
    int kA = half * 16;
    int kB = half * 16;
    int bcol = tn + l15;

    v8f acc = {};
    for (int kk = 0; kk < K; kk += 32) {
        v16bf a, b;
        const float* ap = A + arow * K + kk + kA;
        #pragma unroll
        for (int i = 0; i < 8; ++i) {
            a[i]     = (__bf16)ap[i];
            a[i + 8] = (__bf16)ap[8 + i];
        }
        const float* bp = Bm + (long long)(kk + kB) * Nc + bcol;
        #pragma unroll
        for (int j = 0; j < 16; ++j) b[j] = (__bf16)bp[(long long)j * Nc];
        acc = __builtin_amdgcn_wmma_f32_16x16x32_bf16(
            false, a, false, b, (short)0, acc, false, false);
    }

    int col = tn + l15;
    float bv = bias ? bias[col] : 0.0f;
    #pragma unroll
    for (int r = 0; r < 8; ++r) {
        long long row = tm + half * 8 + r;
        float v = acc[r] + bv;
        if (relu) v = fmaxf(v, 0.0f);
        C[row * (long long)ldc + col] = v;
    }
}

// ---------------------------------------------------------------------------

static inline long long cdiv(long long a, long long b) { return (a + b - 1) / b; }

extern "C" void kernel_launch(void* const* d_in, const int* in_sizes, int n_in,
                              void* d_out, int out_size, void* d_ws, size_t ws_size,
                              hipStream_t stream) {
    // ---- inputs (setup_inputs order) ----
    const float* x    = (const float*)d_in[0];
    const int*   eidx = (const int*)d_in[1];
    const int*   batch= (const int*)d_in[2];
    const float* sfeat= (const float*)d_in[3];
    const float* tfeat= (const float*)d_in[4];
    const int*   tyid = (const int*)d_in[5];
    const float* tyemb= (const float*)d_in[6];
    const float* Ws1=(const float*)d_in[7],  *bs1=(const float*)d_in[8];
    const float* Ws2=(const float*)d_in[9],  *bs2=(const float*)d_in[10];
    const float* Wc1=(const float*)d_in[11], *bc1=(const float*)d_in[12];
    const float* Wc2=(const float*)d_in[13], *bc2=(const float*)d_in[14];
    const float* Wc3=(const float*)d_in[15], *bc3=(const float*)d_in[16];
    const float* Wd1=(const float*)d_in[17], *bd1=(const float*)d_in[18];
    const float* Wd2=(const float*)d_in[19], *bd2=(const float*)d_in[20];
    const float* Wd3=(const float*)d_in[21], *bd3=(const float*)d_in[22];
    const float* Wt1=(const float*)d_in[23], *bt1=(const float*)d_in[24];
    const float* Wt2=(const float*)d_in[25], *bt2=(const float*)d_in[26];
    const float* Wp1=(const float*)d_in[27], *bp1=(const float*)d_in[28];
    const float* Wp2=(const float*)d_in[29], *bp2=(const float*)d_in[30];
    const float* Wf1=(const float*)d_in[31], *bf1=(const float*)d_in[32];
    const float* Wf2=(const float*)d_in[33], *bf2=(const float*)d_in[34];
    const float* Wg =(const float*)d_in[35], *bg =(const float*)d_in[36];
    const float* Wtp=(const float*)d_in[37], *btp=(const float*)d_in[38];

    const int* esrc = eidx;
    const int* edst = eidx + NEDGES;

    // ---- workspace carve-up (floats) ----
    float* ws = (float*)d_ws;
    long long off = 0;
    float* deg   = ws + off; off += NNODES;
    float* dinv  = ws + off; off += NNODES;
    float* ew    = ws + off; off += NEDGES;
    float* P     = ws + off; off += (long long)NNODES * HDIM;   // projection / pattern hidden
    float* AGG   = ws + off; off += (long long)NNODES * HDIM;   // neighbor aggregate
    float* R     = ws + off; off += (long long)NNODES * HDIM;   // residual / pattern out
    float* bufA  = ws + off; off += (long long)NNODES * HDIM;   // x1 / x3
    float* bufB  = ws + off; off += (long long)NNODES * HDIM;   // x2
    float* counts= ws + off; off += NB;
    float* comb  = ws + off; off += (long long)NB * 1280;       // concat buffer
    float* SH    = ws + off; off += (long long)NB * HDIM;       // static hidden
    float* TH    = ws + off; off += (long long)NB * 128;        // temporal hidden
    float* FH    = ws + off; off += (long long)NB * 512;        // fusion hidden

    float* out_fused = (float*)d_out;                 // [64,256]
    float* out_group = out_fused + NB * HDIM;         // [64,20]
    float* out_time  = out_group + NB * NG;           // [64]

    const int TB = 256;
    const long long NH = (long long)NNODES * HDIM;
    const long long EH = (long long)NEDGES * HDIM;

    // small-M generic WMMA GEMM
    auto gemm = [&](const float* A_, const float* B_, const float* bias_, float* C_,
                    int M_, int K_, int Nc_, int ldc_, int relu_) {
        int ntiles = (M_ >> 4) * (Nc_ >> 4);
        int blocks = (int)cdiv(ntiles, TB / 32);
        k_gemm_wmma<<<blocks, TB, 0, stream>>>(A_, B_, bias_, C_, M_, K_, Nc_, ldc_, relu_);
    };

    // large-M strip GEMM (K, NC compile time) with LDS-staged weights
    int stripBlocks = (int)cdiv(NNODES / 16, TB / 32);
    auto gemmN_128_256 = [&](const float* A_, const float* B_, const float* bias_,
                             float* C_, int relu_) {
        k_gemm_strip<128, 256><<<stripBlocks, TB, 256 * (128 / 2 + 4) * 4, stream>>>(
            A_, B_, bias_, C_, NNODES, HDIM, relu_);
    };
    auto gemmN_256_256 = [&](const float* A_, const float* B_, const float* bias_,
                             float* C_, int relu_) {
        k_gemm_strip<256, 256><<<stripBlocks, TB, 256 * (256 / 2 + 4) * 4, stream>>>(
            A_, B_, bias_, C_, NNODES, HDIM, relu_);
    };
    auto gemmN_256_128 = [&](const float* A_, const float* B_, const float* bias_,
                             float* C_, int relu_) {
        k_gemm_strip<256, 128><<<stripBlocks, TB, 128 * (256 / 2 + 4) * 4, stream>>>(
            A_, B_, bias_, C_, NNODES, 128, relu_);
    };

    // ---- GCN norm: deg = 1 + indeg, dinv = deg^-0.5, ew = dinv[src]*dinv[dst] ----
    k_fill<<<(int)cdiv(NNODES, TB), TB, 0, stream>>>(deg, 1.0f, NNODES);
    k_degree<<<(int)cdiv(NEDGES, TB), TB, 0, stream>>>(edst, deg, NEDGES);
    k_dinv<<<(int)cdiv(NNODES, TB), TB, 0, stream>>>(deg, dinv, NNODES);
    k_edgew<<<(int)cdiv(NEDGES, TB), TB, 0, stream>>>(esrc, edst, dinv, ew, NEDGES);

    // ---- layer 1: x[N,128] -> bufA ----
    gemmN_128_256(x, Wc1, nullptr, P, 0);
    k_fill<<<(int)cdiv(NH, TB), TB, 0, stream>>>(AGG, 0.0f, NH);
    k_scatter<<<(int)cdiv(EH, TB), TB, 0, stream>>>(esrc, edst, ew, P, AGG, EH);
    gemmN_128_256(x, Wd1, bd1, R, 0);
    k_combine<<<(int)cdiv(NH, TB), TB, 0, stream>>>(AGG, P, R, deg, bc1, bufA, NH);

    // ---- layer 2: bufA -> bufB ----
    gemmN_256_256(bufA, Wc2, nullptr, P, 0);
    k_fill<<<(int)cdiv(NH, TB), TB, 0, stream>>>(AGG, 0.0f, NH);
    k_scatter<<<(int)cdiv(EH, TB), TB, 0, stream>>>(esrc, edst, ew, P, AGG, EH);
    gemmN_256_256(bufA, Wd2, bd2, R, 0);
    k_combine<<<(int)cdiv(NH, TB), TB, 0, stream>>>(AGG, P, R, deg, bc2, bufB, NH);

    // ---- layer 3: bufB -> bufA (x3) ----
    gemmN_256_256(bufB, Wc3, nullptr, P, 0);
    k_fill<<<(int)cdiv(NH, TB), TB, 0, stream>>>(AGG, 0.0f, NH);
    k_scatter<<<(int)cdiv(EH, TB), TB, 0, stream>>>(esrc, edst, ew, P, AGG, EH);
    gemmN_256_256(bufB, Wd3, bd3, R, 0);
    k_combine<<<(int)cdiv(NH, TB), TB, 0, stream>>>(AGG, P, R, deg, bc3, bufA, NH);

    // ---- pattern detector: pat = relu(relu(x3@Wp1+bp1)@Wp2+bp2) -> R ----
    gemmN_256_128(bufA, Wp1, bp1, P, 1);   // P reused as [N,128]
    gemmN_128_256(P, Wp2, bp2, R, 1);

    // ---- pooling into combined[:, 0:256] and [:, 1024:1280] ----
    k_fill<<<(int)cdiv(NB, TB), TB, 0, stream>>>(counts, 0.0f, NB);
    k_fill<<<(int)cdiv((long long)NB * 1280, TB), TB, 0, stream>>>(comb, 0.0f, (long long)NB * 1280);
    k_counts<<<(int)cdiv(NNODES, TB), TB, 0, stream>>>(batch, counts, NNODES);
    k_pool<<<(int)cdiv(NH, TB), TB, 0, stream>>>(batch, bufA, R, comb, NH);
    k_pooldiv<<<(int)cdiv(NB * HDIM, TB), TB, 0, stream>>>(counts, comb, NB * HDIM);

    // ---- static encoder -> combined[:, 256:512] ----
    gemm(sfeat, Ws1, bs1, SH, NB, DS, HDIM, HDIM, 1);
    gemm(SH, Ws2, bs2, comb + 256, NB, HDIM, HDIM, 1280, 0);

    // ---- temporal encoder -> combined[:, 512:768] ----
    k_temporal1<<<(int)cdiv(NB * 128, TB), TB, 0, stream>>>(tfeat, Wt1, bt1, TH, NB * 128);
    gemm(TH, Wt2, bt2, comb + 512, NB, 128, HDIM, 1280, 0);

    // ---- type embedding -> combined[:, 768:1024] ----
    k_typegather<<<(int)cdiv(NB * HDIM, TB), TB, 0, stream>>>(tyid, tyemb, comb, NB * HDIM);

    // ---- fusion: fused = relu(combined@Wf1+bf1)@Wf2+bf2 -> d_out[0:16384] ----
    gemm(comb, Wf1, bf1, FH, NB, 1280, 512, 512, 1);
    gemm(FH, Wf2, bf2, out_fused, NB, 512, HDIM, HDIM, 0);

    // ---- heads ----
    k_grouphead<<<(int)cdiv(NB * NG, TB), TB, 0, stream>>>(out_fused, Wg, bg, out_group, NB * NG);
    k_timehead<<<1, NB, 0, stream>>>(out_fused, Wtp, btp, out_time, NB);
}
